// RelativeMultiHeadSelfAttention_64604898066472
// MI455X (gfx1250) — compile-verified, matrix operand order checked by reading, not run on hardware
//
#include <hip/hip_runtime.h>
#include <hip/hip_bf16.h>

typedef _Float16 f16;
typedef __attribute__((ext_vector_type(16))) _Float16 v16h;
typedef __attribute__((ext_vector_type(2)))  _Float16 h2;
typedef __attribute__((ext_vector_type(4)))  float    f4;
typedef __attribute__((ext_vector_type(8)))  float    v8f;

#define NB    16      // batch
#define SS    1024    // tokens
#define CC    256     // channels
#define NH    8       // heads
#define HD    32      // head dim
#define DIMM  256     // NH*HD
#define SCALE_F 0.0625f   // DIM^-0.5 = 1/16

#define WMMA(a,b,c) __builtin_amdgcn_wmma_f32_16x16x32_f16(false,(a),false,(b),(short)0,(c),false,false)

// ---------------------------------------------------------------------------
// Fragment loader per CDNA5 ISA §7.12.2 (wave32).
// 16-bit A/B matrix 16x32: r = lane&15 selects the row; VGPR v holds the
// K-pair k = (v>>2)*16 + (lane>>4)*8 + (v&3)*2. Per lane this is two
// contiguous 16-byte chunks -> two global_load_b128.
// ---------------------------------------------------------------------------
__device__ __forceinline__ v16h frag_ld_f16(const f16* __restrict__ src, int ld) {
  const int lane = threadIdx.x & 31;
  const int r = lane & 15, hi = lane >> 4;
  v16h f;
#pragma unroll
  for (int v = 0; v < 8; ++v) {
    const int k = ((v >> 2) << 4) + (hi << 3) + ((v & 3) << 1);
    const h2 p = *(const h2*)(src + (size_t)r * ld + k);
    f[2 * v]     = p.x;
    f[2 * v + 1] = p.y;
  }
  return f;
}

// ---------------------------------------------------------------------------
// Kernel 0: one-shot fp32 -> f16 conversion (x, w_qkv, w_merge).
// ---------------------------------------------------------------------------
__global__ void rmhsa_cvt_f16(const float* __restrict__ src,
                              f16* __restrict__ dst, int n4) {
  const int i = blockIdx.x * blockDim.x + threadIdx.x;
  if (i >= n4) return;
  const f4 v = ((const f4*)src)[i];
  h2 a, b;
  a.x = (f16)v.x; a.y = (f16)v.y;
  b.x = (f16)v.z; b.y = (f16)v.w;
  ((h2*)dst)[2 * i]     = a;
  ((h2*)dst)[2 * i + 1] = b;
}

// ---------------------------------------------------------------------------
// Kernel 1: gather relative-position bias, transposed: biasT[k*S + q].
// ---------------------------------------------------------------------------
__global__ void rmhsa_bias_gather(const float* __restrict__ rel_table,
                                  const int* __restrict__ rel_index,
                                  float* __restrict__ biasT) {
  const int idx = blockIdx.x * blockDim.x + threadIdx.x;   // over S*S
  const int k = idx >> 10;
  const int q = idx & (SS - 1);
  biasT[idx] = rel_table[rel_index[q * SS + k] * NH];      // column 0 only
}

// ---------------------------------------------------------------------------
// Kernel 2: QKV projection. qkv = x @ w_qkv^T + b_qkv, all-f16 fragments.
// One wave -> 16x64 output (A fragment reused for 4 WMMAs per k-step).
// Writes f16: Q[bh][s][d], K[bh][s][d], VT[bh][d][s].
// ---------------------------------------------------------------------------
__global__ void rmhsa_qkv_proj(const f16* __restrict__ xh,
                               const f16* __restrict__ wqh,
                               const float* __restrict__ b_qkv,
                               f16* __restrict__ Q, f16* __restrict__ K,
                               f16* __restrict__ VT) {
  const int wid = blockIdx.x * (blockDim.x >> 5) + (threadIdx.x >> 5);
  const int nt4 = wid % 12;           // 768/64 col groups
  const int mt  = wid / 12;           // 16384/16 row tiles
  const int nbase = nt4 * 64;
  const f16* xrow = xh + (size_t)mt * 16 * CC;

  v8f c[4] = {{}, {}, {}, {}};
#pragma unroll
  for (int k = 0; k < CC; k += 32) {
    const v16h a = frag_ld_f16(xrow + k, CC);
#pragma unroll
    for (int t = 0; t < 4; ++t) {
      const v16h b = frag_ld_f16(wqh + (size_t)(nbase + t * 16) * CC + k, CC);
      c[t] = WMMA(a, b, c[t]);
    }
  }

  const int lane = threadIdx.x & 31;
  const int n = lane & 15, hi = lane >> 4;
#pragma unroll
  for (int t = 0; t < 4; ++t) {
    const int nout = nbase + t * 16 + n;
    const float bias = b_qkv[nout];
    const int part = nout >> 8;               // uniform per t (64 | 256)
    const int nn = nout & 255;
    const int h = nn >> 5, d = nn & 31;
#pragma unroll
    for (int j = 0; j < 8; ++j) {
      const int mg = mt * 16 + j + hi * 8;    // global row in [B*S]
      const int b_ = mg >> 10, s = mg & (SS - 1);
      const size_t bh = (size_t)(b_ * NH + h);
      const f16 val = (f16)(c[t][j] + bias);
      if (part == 0)      Q [(bh * SS + s) * HD + d] = val;
      else if (part == 1) K [(bh * SS + s) * HD + d] = val;
      else                VT[(bh * HD + d) * SS + s] = val;   // transposed V
    }
  }
}

// ---------------------------------------------------------------------------
// Kernel 3: fused flash attention. One wave owns one (b,h,16-query) tile,
// streams K/V in blocks of 32 keys, online softmax in fp32, P->f16 via
// wave-private LDS to convert C-layout -> A-fragment layout.
// ---------------------------------------------------------------------------
__global__ void rmhsa_flash_attn(const f16* __restrict__ Q,
                                 const f16* __restrict__ K,
                                 const f16* __restrict__ VT,
                                 const float* __restrict__ biasT,
                                 f16* __restrict__ AO) {
  __shared__ _Float16 plds[8 * 16 * 32];      // 1 KB per wave, 8 waves/block
  const int wslot = threadIdx.x >> 5;
  f16* pl = (f16*)plds + wslot * 512;

  const int wid = blockIdx.x * (blockDim.x >> 5) + wslot;   // 0..8191
  const int qt = wid & 63;                    // 64 query tiles
  const int bh = wid >> 6;                    // 0..127  (b*8+h)
  const int h = bh & 7, b_ = bh >> 3;
  const int q0 = qt * 16;

  const f16* Qbh  = Q  + (size_t)bh * SS * HD;
  const f16* Kbh  = K  + (size_t)bh * SS * HD;
  const f16* VTbh = VT + (size_t)bh * HD * SS;

  const int lane = threadIdx.x & 31;
  const int n = lane & 15, hi = lane >> 4;

  const v16h qf = frag_ld_f16(Qbh + (size_t)q0 * HD, HD);

  v8f o0 = {}, o1 = {};
  float rm[8], rl[8];
#pragma unroll
  for (int j = 0; j < 8; ++j) { rm[j] = -3.0e38f; rl[j] = 0.0f; }

  for (int kb = 0; kb < SS; kb += 32) {
    if (kb + 32 < SS) {                       // uniform speculative prefetch
      __builtin_prefetch(Kbh + (size_t)(kb + 32) * HD, 0, 0);
      __builtin_prefetch(VTbh + kb + 32, 0, 0);
    }
    // scores: two 16x16 tiles covering 32 keys (K-dim = HEAD_DIM = 32)
    v16h kf0 = frag_ld_f16(Kbh + (size_t)kb * HD, HD);
    v16h kf1 = frag_ld_f16(Kbh + (size_t)(kb + 16) * HD, HD);
    v8f c0 = {}, c1 = {};
    c0 = WMMA(qf, kf0, c0);
    c1 = WMMA(qf, kf1, c1);

    // scale + transposed bias (8 contiguous floats per lane per tile)
    const float* bp0 = biasT + (size_t)(kb + n)      * SS + q0 + hi * 8;
    const float* bp1 = biasT + (size_t)(kb + 16 + n) * SS + q0 + hi * 8;
    float s0[8], s1[8];
#pragma unroll
    for (int j = 0; j < 8; ++j) {
      s0[j] = c0[j] * SCALE_F + bp0[j];
      s1[j] = c1[j] * SCALE_F + bp1[j];
    }

    // row max across the 32 keys: lane-local max then 16-lane xor reduce
    float mx[8];
#pragma unroll
    for (int j = 0; j < 8; ++j) mx[j] = fmaxf(s0[j], s1[j]);
#pragma unroll
    for (int m = 1; m < 16; m <<= 1)
#pragma unroll
      for (int j = 0; j < 8; ++j) mx[j] = fmaxf(mx[j], __shfl_xor(mx[j], m, 32));

    float corr[8];
#pragma unroll
    for (int j = 0; j < 8; ++j) {
      const float nm = fmaxf(rm[j], mx[j]);
      corr[j] = __expf(rm[j] - nm);
      rm[j] = nm;
    }

    float p0[8], p1[8], sum[8];
#pragma unroll
    for (int j = 0; j < 8; ++j) {
      p0[j] = __expf(s0[j] - rm[j]);
      p1[j] = __expf(s1[j] - rm[j]);
      sum[j] = p0[j] + p1[j];
    }
#pragma unroll
    for (int m = 1; m < 16; m <<= 1)
#pragma unroll
      for (int j = 0; j < 8; ++j) sum[j] += __shfl_xor(sum[j], m, 32);

#pragma unroll
    for (int j = 0; j < 8; ++j) {
      rl[j] = rl[j] * corr[j] + sum[j];
      o0[j] *= corr[j];
      o1[j] *= corr[j];
    }

    // C-layout -> A-fragment layout via wave-private LDS (in-order DS pipe)
#pragma unroll
    for (int j = 0; j < 8; ++j) {
      const int row = j + hi * 8;
      pl[row * 32 + n]      = (f16)p0[j];
      pl[row * 32 + n + 16] = (f16)p1[j];
    }
    v16h pf = frag_ld_f16(pl, 32);

    // O += P (16x32 keys) x V (32 keys x 32 dims), dims split into 2 N-tiles
    v16h vf0 = frag_ld_f16(VTbh + kb, SS);                   // dims 0..15
    v16h vf1 = frag_ld_f16(VTbh + (size_t)16 * SS + kb, SS); // dims 16..31
    o0 = WMMA(pf, vf0, o0);
    o1 = WMMA(pf, vf1, o1);
  }

  // normalize and write AO[b][s][h*32 + d] as f16 for the merge GEMM
#pragma unroll
  for (int j = 0; j < 8; ++j) {
    const int row = j + hi * 8;
    const int s = q0 + row;
    const float inv = 1.0f / rl[j];
    f16* dst = AO + ((size_t)b_ * SS + s) * DIMM + h * HD;
    dst[n]      = (f16)(o0[j] * inv);
    dst[n + 16] = (f16)(o1[j] * inv);
  }
}

// ---------------------------------------------------------------------------
// Kernel 4: merge projection. out = AO @ w_merge^T + b_merge (fp32 out).
// One wave -> 16x64 output (A fragment reused for 4 WMMAs per k-step).
// ---------------------------------------------------------------------------
__global__ void rmhsa_merge_proj(const f16* __restrict__ AO,
                                 const f16* __restrict__ wmh,
                                 const float* __restrict__ b_merge,
                                 float* __restrict__ out) {
  const int wid = blockIdx.x * (blockDim.x >> 5) + (threadIdx.x >> 5);
  const int nt4 = wid & 3;            // 256/64 col groups
  const int mt  = wid >> 2;           // 16384/16 row tiles
  const int nbase = nt4 * 64;
  const f16* arow = AO + (size_t)mt * 16 * DIMM;

  v8f c[4] = {{}, {}, {}, {}};
#pragma unroll
  for (int k = 0; k < DIMM; k += 32) {
    const v16h a = frag_ld_f16(arow + k, DIMM);
#pragma unroll
    for (int t = 0; t < 4; ++t) {
      const v16h b = frag_ld_f16(wmh + (size_t)(nbase + t * 16) * DIMM + k, DIMM);
      c[t] = WMMA(a, b, c[t]);
    }
  }

  const int lane = threadIdx.x & 31;
  const int n = lane & 15, hi = lane >> 4;
#pragma unroll
  for (int t = 0; t < 4; ++t) {
    const int nout = nbase + t * 16 + n;
    const float bias = b_merge[nout];
#pragma unroll
    for (int j = 0; j < 8; ++j) {
      const int mg = mt * 16 + j + hi * 8;
      out[(size_t)mg * CC + nout] = c[t][j] + bias;
    }
  }
}

// ---------------------------------------------------------------------------
extern "C" void kernel_launch(void* const* d_in, const int* in_sizes, int n_in,
                              void* d_out, int out_size, void* d_ws, size_t ws_size,
                              hipStream_t stream) {
  const float* x         = (const float*)d_in[0];
  const float* w_qkv     = (const float*)d_in[1];
  const float* b_qkv     = (const float*)d_in[2];
  const float* w_merge   = (const float*)d_in[3];
  const float* b_merge   = (const float*)d_in[4];
  const float* rel_table = (const float*)d_in[5];
  const int*   rel_index = (const int*)d_in[6];
  float* out = (float*)d_out;

  char* ws = (char*)d_ws;
  const size_t MB = 1024 * 1024;
  f16*   Q     = (f16*)(ws + 0 * MB);          // 8 MB  [B*H][S][32]
  f16*   K     = (f16*)(ws + 8 * MB);          // 8 MB  [B*H][S][32]
  f16*   VT    = (f16*)(ws + 16 * MB);         // 8 MB  [B*H][32][S]
  f16*   XH    = (f16*)(ws + 24 * MB);         // 8 MB  f16 x (dead after QKV)
  f16*   AO    = (f16*)(ws + 24 * MB);         // 8 MB  [B*S][256] (reuses XH)
  float* biasT = (float*)(ws + 32 * MB);       // 4 MB  [S][S] transposed
  f16*   WQH   = (f16*)(ws + 36 * MB);         // 384 KB f16 w_qkv
  f16*   WMH   = (f16*)(ws + 36 * MB + 512 * 1024); // 128 KB f16 w_merge

  // one-shot conversions (vec4 per thread)
  rmhsa_cvt_f16<<<(NB * SS * CC / 4 + 255) / 256, 256, 0, stream>>>(x, XH, NB * SS * CC / 4);
  rmhsa_cvt_f16<<<(3 * DIMM * CC / 4 + 255) / 256, 256, 0, stream>>>(w_qkv, WQH, 3 * DIMM * CC / 4);
  rmhsa_cvt_f16<<<(CC * DIMM / 4 + 255) / 256, 256, 0, stream>>>(w_merge, WMH, CC * DIMM / 4);

  rmhsa_bias_gather<<<(SS * SS) / 256, 256, 0, stream>>>(rel_table, rel_index, biasT);
  rmhsa_qkv_proj  <<<1536, 256, 0, stream>>>(XH, WQH, b_qkv, Q, K, VT);    // 12288 waves
  rmhsa_flash_attn<<<1024, 256, 0, stream>>>(Q, K, VT, biasT, AO);         //  8192 waves
  rmhsa_merge_proj<<<512, 256, 0, stream>>>(AO, WMH, b_merge, out);        //  4096 waves
}
